// Transducer_75307956568725
// MI455X (gfx1250) — matrix operand order repaired and checked by reading, not compiled, not run on hardware
//
#include <hip/hip_runtime.h>

// MI455X / gfx1250, wave32. bf16 WMMA (V_WMMA_F32_16X16X32_BF16) everywhere.
// All GEMM operands pre-packed into WMMA fragment order so hot loops are
// branch-free b128 loads + v_wmma only.  B fragments for a k-step are loaded
// as a batch before the WMMA burst so the backend can stagger s_wait_loadcnt
// and overlap loads with matrix math.

typedef __attribute__((ext_vector_type(16))) __bf16 v16bf;
typedef __attribute__((ext_vector_type(8)))  float  v8f;

#define B_   4
#define T_   256
#define U_   64
#define V_   1024
#define F_   80
#define H_   512
#define J_   512
#define U1_  65
#define NEGF (-1e30f)

__device__ __forceinline__ __bf16 f2bf(float x) { return (__bf16)x; }

__device__ __forceinline__ float laddexp(float x, float y) {
  float m = fmaxf(x, y);
  float n = fminf(x, y);
  return m + log1pf(__expf(n - m));
}

// ---------------------------------------------------------------------------
// Pack A (Msrc x Ksrc f32 row-major) -> A-fragment order bf16:
// out[mt][ks][lane][i]: row = mt*16 + (lane&15),
// k = ks*32 + (lane>>4)*8 + (i<8 ? i : 8+i).  Zero-pad beyond Msrc/Ksrc.
// ---------------------------------------------------------------------------
__global__ void pack_a_kernel(const float* __restrict__ src,
                              __bf16* __restrict__ dst,
                              int Msrc, int Ksrc, int KS, int total) {
  int idx = blockIdx.x * blockDim.x + threadIdx.x;
  if (idx >= total) return;
  int i    =  idx       & 15;
  int lane = (idx >> 4) & 31;
  int rest =  idx >> 9;
  int ks   = rest % KS;
  int mt   = rest / KS;
  int row  = mt * 16 + (lane & 15);
  int k    = ks * 32 + ((lane >> 4) * 8) + ((i < 8) ? i : (8 + i));
  float v  = (row < Msrc && k < Ksrc) ? src[(size_t)row * Ksrc + k] : 0.f;
  dst[idx] = f2bf(v);
}

// ---------------------------------------------------------------------------
// Pack B (Ksrc x N f32 row-major) -> B-fragment order bf16:
// out[nt][ks][lane][i]: col = nt*16 + (lane&15), k = ks*32 + (lane>>4)*8 + (i<8 ? i : 8+i).
// ---------------------------------------------------------------------------
__global__ void pack_b_kernel(const float* __restrict__ src,
                              __bf16* __restrict__ dst,
                              int Ksrc, int N, int KS, int total) {
  int idx = blockIdx.x * blockDim.x + threadIdx.x;
  if (idx >= total) return;
  int i    =  idx       & 15;
  int lane = (idx >> 4) & 31;
  int rest =  idx >> 9;
  int ks   = rest % KS;
  int nt   = rest / KS;
  int col  = nt * 16 + (lane & 15);
  int k    = ks * 32 + ((lane >> 4) * 8) + ((i < 8) ? i : (8 + i));
  float v  = (k < Ksrc) ? src[(size_t)k * N + col] : 0.f;
  dst[idx] = f2bf(v);
}

// ---------------------------------------------------------------------------
// dec[b*(U+1)+u][:] = emb[padded_target]  (blank=0 prepended)
// ---------------------------------------------------------------------------
__global__ void gather_dec_kernel(const float* __restrict__ emb,
                                  const int* __restrict__ targets,
                                  float* __restrict__ dec) {
  int idx = blockIdx.x * blockDim.x + threadIdx.x;
  if (idx >= B_ * U1_ * 512) return;
  int row = idx >> 9;
  int k   = idx & 511;
  int b   = row / U1_;
  int u   = row - b * U1_;
  int tok = (u == 0) ? 0 : targets[b * U_ + (u - 1)];
  dec[idx] = emb[(size_t)tok * 512 + k];
}

// ---------------------------------------------------------------------------
// Fully packed GEMM.  Block = 256 threads = 8 waves covering 16 x (8*NTW*16)
// of C.  Wave w owns NTW 16x16 tiles, reusing one A fragment per k-step.
// No LDS, no guards: batched b128 loads then a v_wmma burst.
// ---------------------------------------------------------------------------
template <int KS, int NTW>
__global__ void gemm_packed_kernel(const __bf16* __restrict__ pA,
                                   const __bf16* __restrict__ pB,
                                   float* __restrict__ C, int ldc) {
  const int tid  = threadIdx.x;
  const int lane = tid & 31;
  const int w    = tid >> 5;
  const int mt   = blockIdx.x;

  v8f acc[NTW];
  v8f zero = {0.f, 0.f, 0.f, 0.f, 0.f, 0.f, 0.f, 0.f};
  #pragma unroll
  for (int n = 0; n < NTW; ++n) acc[n] = zero;

  const __bf16* aBase = pA + ((size_t)mt * KS * 32 + lane) * 16;
  const __bf16* bBase = pB + ((size_t)(w * NTW) * KS * 32 + lane) * 16;

  for (int ks = 0; ks < KS; ++ks) {
    v16bf a = *(const v16bf*)(aBase + (size_t)ks * 512);
    v16bf bw[NTW];
    #pragma unroll
    for (int n = 0; n < NTW; ++n) {
      const __bf16* bp = bBase + ((size_t)n * KS + ks) * 512;
      __builtin_prefetch(bp + 512, 0, 1);
      bw[n] = *(const v16bf*)bp;
    }
    #pragma unroll
    for (int n = 0; n < NTW; ++n)
      acc[n] = __builtin_amdgcn_wmma_f32_16x16x32_bf16(
          false, a, false, bw[n], (short)0, acc[n], false, false);
  }

  #pragma unroll
  for (int n = 0; n < NTW; ++n) {
    int col = (w * NTW + n) * 16 + (lane & 15);
    #pragma unroll
    for (int r = 0; r < 8; ++r) {
      int row = mt * 16 + r + 8 * (lane >> 4);          // C/D layout (ISA 7.12.2)
      C[(size_t)row * ldc + col] = acc[n][r];
    }
  }
}

// ---------------------------------------------------------------------------
// Joint + online log-softmax.  One block = 16 rows of flattened (b,t,u).
// Thread (row, seg) computes k-step seg (32 j's) of its row: float4 loads,
// tanh, then four contiguous 8x bf16 runs -> ds_store_b128 into A-frag LDS.
// 8 waves x 8 N-tiles each cover V=1024; K=512 -> 16 WMMA steps with A-frag
// reuse across 8 accumulators; per k-step: 16 batched global_load_b128 then
// 8 v_wmma.  Online LSE via shfl_xor butterflies + cross-wave LDS reduction;
// blank/label logits captured by shfl.
// ---------------------------------------------------------------------------
__global__ void joint_kernel(const float* __restrict__ e,
                             const float* __restrict__ d,
                             const float* __restrict__ b_j,
                             const __bf16* __restrict__ pack,
                             const float* __restrict__ b_out,
                             const int* __restrict__ targets,
                             const int* __restrict__ tgt_lens,
                             float* __restrict__ blankbuf,
                             float* __restrict__ lblbuf) {
  __shared__ alignas(32) __bf16 sA[16][32][16];   // [ks][lane][i], 16 KB
  __shared__ int   s_c[16], s_u[16], s_bt[16], s_tl[16];
  __shared__ float s_rm[8][16], s_rs[8][16];
  __shared__ float s_blank[16], s_lbl[16];

  const int tid  = threadIdx.x;
  const int lane = tid & 31;
  const int w    = tid >> 5;
  const int g0   = blockIdx.x * 16;

  // ---- stage h = tanh(e + d + b_j) in A-fragment order ----
  {
    int row = tid >> 4;
    int seg = tid & 15;        // == ks handled by this thread
    int g   = g0 + row;
    int b   = g / (T_ * U1_);
    int rem = g - b * (T_ * U1_);
    int t   = rem / U1_;
    int u   = rem - t * U1_;
    if (seg == 0) {
      s_u[row]  = u;
      s_bt[row] = b * T_ + t;
      s_tl[row] = tgt_lens[b];
      s_c[row]  = (u < U_) ? targets[b * U_ + u] : 0;
      s_blank[row] = 0.f;
      s_lbl[row]   = 0.f;
    }
    const float4* e4 = (const float4*)(e + (size_t)(b * T_ + t) * J_ + seg * 32);
    const float4* d4 = (const float4*)(d + (size_t)(b * U1_ + u) * J_ + seg * 32);
    const float4* j4 = (const float4*)(b_j + seg * 32);
    float h32[32];
    #pragma unroll
    for (int q = 0; q < 8; ++q) {
      float4 ev = e4[q], dv = d4[q], bv = j4[q];
      h32[q * 4 + 0] = tanhf(ev.x + dv.x + bv.x);
      h32[q * 4 + 1] = tanhf(ev.y + dv.y + bv.y);
      h32[q * 4 + 2] = tanhf(ev.z + dv.z + bv.z);
      h32[q * 4 + 3] = tanhf(ev.w + dv.w + bv.w);
    }
    // runs: kk 0..7 -> [row][0..7], 8..15 -> [row+16][0..7],
    //       16..23 -> [row][8..15], 24..31 -> [row+16][8..15]
    #pragma unroll
    for (int run = 0; run < 4; ++run) {
      alignas(16) __bf16 tmp[8];
      #pragma unroll
      for (int z = 0; z < 8; ++z) tmp[z] = f2bf(h32[run * 8 + z]);
      int dl = row + ((run & 1) ? 16 : 0);
      int di = (run >> 1) * 8;
      *(uint4*)&sA[seg][dl][di] = *(const uint4*)tmp;
    }
  }
  __syncthreads();

  // ---- 16x16x32 bf16 WMMA: 16 K-steps x 8 N-tiles per wave ----
  v8f acc[8];
  v8f zero = {0.f, 0.f, 0.f, 0.f, 0.f, 0.f, 0.f, 0.f};
  #pragma unroll
  for (int nt = 0; nt < 8; ++nt) acc[nt] = zero;

  const int ntbase = w * 8;
  const __bf16* pBase = pack + ((size_t)ntbase * 16 * 32 + lane) * 16;
  for (int ks = 0; ks < 16; ++ks) {
    v16bf a = *(const v16bf*)&sA[ks][lane][0];
    v16bf bw[8];
    #pragma unroll
    for (int nt = 0; nt < 8; ++nt) {
      const __bf16* bp = pBase + ((size_t)nt * 16 + ks) * 512;
      __builtin_prefetch(bp + 512, 0, 1);          // next ks, same nt
      bw[nt] = *(const v16bf*)bp;
    }
    #pragma unroll
    for (int nt = 0; nt < 8; ++nt)
      acc[nt] = __builtin_amdgcn_wmma_f32_16x16x32_bf16(
          false, a, false, bw[nt], (short)0, acc[nt], false, false);
  }

  // ---- + b_out ----
  #pragma unroll
  for (int nt = 0; nt < 8; ++nt) {
    float bo = b_out[(ntbase + nt) * 16 + (lane & 15)];
    #pragma unroll
    for (int r = 0; r < 8; ++r) acc[nt][r] += bo;
  }

  // ---- per-wave partial max / sum-exp per row (butterflies stay in half) ----
  #pragma unroll
  for (int r = 0; r < 8; ++r) {
    float m = NEGF;
    #pragma unroll
    for (int nt = 0; nt < 8; ++nt) m = fmaxf(m, acc[nt][r]);
    m = fmaxf(m, __shfl_xor(m, 1));
    m = fmaxf(m, __shfl_xor(m, 2));
    m = fmaxf(m, __shfl_xor(m, 4));
    m = fmaxf(m, __shfl_xor(m, 8));
    float s = 0.f;
    #pragma unroll
    for (int nt = 0; nt < 8; ++nt) s += __expf(acc[nt][r] - m);
    s += __shfl_xor(s, 1);
    s += __shfl_xor(s, 2);
    s += __shfl_xor(s, 4);
    s += __shfl_xor(s, 8);
    int row = r + 8 * (lane >> 4);
    if ((lane & 15) == 0) { s_rm[w][row] = m; s_rs[w][row] = s; }
  }

  // ---- capture blank (col 0, wave 0) and label logits ----
  if (w == 0) {
    #pragma unroll
    for (int r = 0; r < 8; ++r) {
      float v0 = __shfl(acc[0][r], 0);
      float v1 = __shfl(acc[0][r], 16);
      if (lane == 0) { s_blank[r] = v0; s_blank[r + 8] = v1; }
    }
  }
  #pragma unroll
  for (int r = 0; r < 8; ++r) {
    #pragma unroll
    for (int half = 0; half < 2; ++half) {
      int row = r + 8 * half;
      int c   = s_c[row];                    // uniform per row across wave
      if ((c >> 7) == w) {
        int ntl = (c >> 4) & 7;
        int src = half * 16 + (c & 15);
        #pragma unroll
        for (int nt = 0; nt < 8; ++nt) {
          if (nt == ntl) {
            float v = __shfl(acc[nt][r], src);
            if (lane == 0) s_lbl[row] = v;
          }
        }
      }
    }
  }
  __syncthreads();

  // ---- cross-wave LSE, write outputs ----
  if (tid < 16) {
    int row = tid;
    float M = NEGF;
    #pragma unroll
    for (int q = 0; q < 8; ++q) M = fmaxf(M, s_rm[q][row]);
    float S = 0.f;
    #pragma unroll
    for (int q = 0; q < 8; ++q) S += s_rs[q][row] * __expf(s_rm[q][row] - M);
    float lse = M + __logf(S);
    int g = g0 + row;
    blankbuf[g] = s_blank[row] - lse;
    int u = s_u[row];
    if (u < U_) {
      float lp = (u < s_tl[row]) ? (s_lbl[row] - lse) : NEGF;
      lblbuf[(size_t)s_bt[row] * U_ + u] = lp;
    }
  }
}

// ---------------------------------------------------------------------------
// RNN-T alpha recurrence (one thread per batch element) + final loss.
// ---------------------------------------------------------------------------
__global__ void dp_kernel(const float* __restrict__ blankbuf,
                          const float* __restrict__ lblbuf,
                          const int* __restrict__ in_lens,
                          const int* __restrict__ tgt_lens,
                          float* __restrict__ out) {
  __shared__ float alpha[B_][U1_];
  __shared__ float afin[B_];
  int tid = threadIdx.x;
  if (tid < B_) {
    int b = tid;
    float* al = alpha[b];
    const float* bl = blankbuf + (size_t)b * T_ * U1_;
    const float* lb = lblbuf   + (size_t)b * T_ * U_;
    al[0] = 0.f;
    for (int u = 1; u <= U_; ++u) al[u] = al[u - 1] + lb[u - 1];   // t=0 cumsum
    int tl = in_lens[b] - 1;
    int ul = tgt_lens[b];
    float fin = (tl == 0) ? (al[ul] + bl[ul]) : 0.f;
    for (int t = 1; t < T_; ++t) {
      const float* blp = bl + (size_t)(t - 1) * U1_;
      const float* lbp = lb + (size_t)t * U_;
      float c = al[0] + blp[0];
      al[0] = c;
      for (int u = 1; u <= U_; ++u) {
        float x = al[u] + blp[u];
        float y = c + lbp[u - 1];
        c = laddexp(x, y);
        al[u] = c;
      }
      if (t == tl) fin = al[ul] + bl[(size_t)t * U1_ + ul];
    }
    afin[b] = fin;
  }
  __syncthreads();
  if (tid == 0)
    out[0] = -0.25f * (afin[0] + afin[1] + afin[2] + afin[3]);
}

// ---------------------------------------------------------------------------
extern "C" void kernel_launch(void* const* d_in, const int* in_sizes, int n_in,
                              void* d_out, int out_size, void* d_ws,
                              size_t ws_size, hipStream_t stream) {
  (void)in_sizes; (void)n_in; (void)out_size; (void)ws_size;
  const float* inputs  = (const float*)d_in[0];
  const float* W_enc   = (const float*)d_in[1];
  const float* emb     = (const float*)d_in[2];
  const float* W_jenc  = (const float*)d_in[3];
  const float* W_jdec  = (const float*)d_in[4];
  const float* b_j     = (const float*)d_in[5];
  const float* W_out   = (const float*)d_in[6];
  const float* b_out   = (const float*)d_in[7];
  const int*   targets = (const int*)d_in[8];
  const int*   in_lens = (const int*)d_in[9];
  const int*   tg_lens = (const int*)d_in[10];

  char* ws = (char*)d_ws;
  size_t off = 0;
  auto wsAlloc = [&](size_t bytes) -> void* {
    void* p = ws + off;
    off = (off + bytes + 255) & ~(size_t)255;
    return p;
  };
  // f32 intermediates
  float*  enc   = (float*)wsAlloc((size_t)1024 * 512 * 4);
  float*  ebuf  = (float*)wsAlloc((size_t)1024 * 512 * 4);
  float*  dec   = (float*)wsAlloc((size_t)272 * 512 * 4);
  float*  dbuf  = (float*)wsAlloc((size_t)272 * 512 * 4);
  // packed bf16 operands
  __bf16* pAin  = (__bf16*)wsAlloc((size_t)64 * 3  * 512 * 2);  // inputs  M=1024 K=96
  __bf16* pBenc = (__bf16*)wsAlloc((size_t)32 * 3  * 512 * 2);  // W_enc   K=96  N=512
  __bf16* pAenc = (__bf16*)wsAlloc((size_t)64 * 16 * 512 * 2);  // enc     M=1024 K=512
  __bf16* pBje  = (__bf16*)wsAlloc((size_t)32 * 16 * 512 * 2);  // W_jenc
  __bf16* pAdec = (__bf16*)wsAlloc((size_t)17 * 16 * 512 * 2);  // dec     M=272 K=512
  __bf16* pBjd  = (__bf16*)wsAlloc((size_t)32 * 16 * 512 * 2);  // W_jdec
  __bf16* pWout = (__bf16*)wsAlloc((size_t)64 * 16 * 512 * 2);  // W_out   K=512 N=1024
  // small outputs
  float*  blankbuf = (float*)wsAlloc((size_t)B_ * T_ * U1_ * 4);
  float*  lblbuf   = (float*)wsAlloc((size_t)B_ * T_ * U_ * 4);

  const int BLK = 256;
  auto nb = [](int total, int blk) { return (total + blk - 1) / blk; };

  // packs that don't depend on GEMM outputs
  int tAin  = 64 * 3 * 512;
  int tBenc = 32 * 3 * 512;
  int tBje  = 32 * 16 * 512;
  int tBjd  = 32 * 16 * 512;
  int tWout = 64 * 16 * 512;
  pack_a_kernel<<<nb(tAin, BLK), BLK, 0, stream>>>(inputs, pAin, 1024, 80, 3, tAin);
  pack_b_kernel<<<nb(tBenc, BLK), BLK, 0, stream>>>(W_enc, pBenc, 80, 512, 3, tBenc);
  pack_b_kernel<<<nb(tBje, BLK), BLK, 0, stream>>>(W_jenc, pBje, 512, 512, 16, tBje);
  pack_b_kernel<<<nb(tBjd, BLK), BLK, 0, stream>>>(W_jdec, pBjd, 512, 512, 16, tBjd);
  pack_b_kernel<<<nb(tWout, BLK), BLK, 0, stream>>>(W_out, pWout, 512, 1024, 16, tWout);
  gather_dec_kernel<<<nb(B_ * U1_ * 512, BLK), BLK, 0, stream>>>(emb, targets, dec);

  // enc = inputs @ W_enc
  gemm_packed_kernel<3, 4><<<dim3(64), dim3(BLK), 0, stream>>>(pAin, pBenc, enc, 512);
  int tAenc = 64 * 16 * 512;
  pack_a_kernel<<<nb(tAenc, BLK), BLK, 0, stream>>>(enc, pAenc, 1024, 512, 16, tAenc);
  // e = enc @ W_jenc
  gemm_packed_kernel<16, 4><<<dim3(64), dim3(BLK), 0, stream>>>(pAenc, pBje, ebuf, 512);
  // d = dec @ W_jdec
  int tAdec = 17 * 16 * 512;
  pack_a_kernel<<<nb(tAdec, BLK), BLK, 0, stream>>>(dec, pAdec, 260, 512, 16, tAdec);
  gemm_packed_kernel<16, 4><<<dim3(17), dim3(BLK), 0, stream>>>(pAdec, pBjd, dbuf, 512);

  // joint + online log-softmax (only blank/label log-probs materialized)
  joint_kernel<<<dim3(4160), dim3(BLK), 0, stream>>>(
      ebuf, dbuf, b_j, pWout, b_out, targets, tg_lens, blankbuf, lblbuf);

  // RNN-T DP + loss
  dp_kernel<<<dim3(1), dim3(64), 0, stream>>>(blankbuf, lblbuf, in_lens,
                                              tg_lens, (float*)d_out);
}
